// PerLeadTimeMLP_1236950581948
// MI455X (gfx1250) — matrix operand order repaired and checked by reading, not compiled, not run on hardware
//
#include <hip/hip_runtime.h>

typedef __attribute__((ext_vector_type(16))) _Float16 v16h;
typedef __attribute__((ext_vector_type(8)))  float    v8f;
typedef __attribute__((ext_vector_type(4)))  float    v4f;

#define kT 42
#define kD 2
#define kH 64
#define M_ITERS 16
#define WAVES 8
#define ROWS_PER_WAVE (16 * M_ITERS)            // 256 rows per wave
#define ROWS_PER_BLOCK (WAVES * ROWS_PER_WAVE)  // 2048 rows per block

__global__ __launch_bounds__(256)
void perlead_mlp_wmma_t(const float* __restrict__ x,
                        const float* __restrict__ W1, const float* __restrict__ b1,
                        const float* __restrict__ W2, const float* __restrict__ b2,
                        const float* __restrict__ W3, const float* __restrict__ b3,
                        float* __restrict__ out)
{
  __shared__ float sW3[2][kH];                  // W3 de-interleaved: [d][hidden]

  const int t   = blockIdx.x;                   // lead 0..41
  const int tid = threadIdx.x;

  if (tid < kH) {
    sW3[0][tid] = W3[(t * kH + tid) * kD + 0];
    sW3[1][tid] = W3[(t * kH + tid) * kD + 1];
  }
  __syncthreads();

  const int lane = tid & 31;
  const int wave = tid >> 5;
  const int m    = lane & 15;                   // batch row within tile (= N column)
  const int half = lane >> 4;
  const int rowWave = blockIdx.y * ROWS_PER_BLOCK + wave * ROWS_PER_WAVE;

  // ---- Layer-1 weights in B-operand layout: K contiguous, k = 32s + 16*half + e ----
  v16h w1r0[2], w1r1[2], b1p[2];
  #pragma unroll
  for (int s = 0; s < 2; ++s) {
    #pragma unroll
    for (int e = 0; e < 16; ++e) {
      const int k = 32 * s + 16 * half + e;
      w1r0[s][e] = (_Float16)W1[(t * kD + 0) * kH + k];
      w1r1[s][e] = (_Float16)W1[(t * kD + 1) * kH + k];
      b1p[s][e]  = (_Float16)b1[t * kH + k];
    }
  }

  // ---- W2^T in A-operand layout: M row = hidden_out = 16j + m,
  //      K = hidden_in = 16*half + e + 32s ----
  v16h aW2[4][2];
  #pragma unroll
  for (int j = 0; j < 4; ++j) {
    #pragma unroll
    for (int s = 0; s < 2; ++s) {
      #pragma unroll
      for (int e = 0; e < 16; ++e) {
        const int kin = 16 * half + e + 32 * s;
        aW2[j][s][e] = (_Float16)W2[(t * kH + kin) * kH + 16 * j + m];
      }
    }
  }

  // ---- b2 as WMMA C-init: C row M = hidden_out = 16j + 8*half + v ----
  v8f b2r[4];
  #pragma unroll
  for (int j = 0; j < 4; ++j) {
    #pragma unroll
    for (int v = 0; v < 8; ++v)
      b2r[j][v] = b2[t * kH + 16 * j + 8 * half + v];
  }

  const float b3v = b3[t * kD + half];
  const float* w3p0 = &sW3[0][8 * half];
  const float* w3p1 = &sW3[1][8 * half];

  for (int it = 0; it < M_ITERS; ++it) {
    const int rowTile = rowWave + it * 16;
    const int row = rowTile + m;                 // both halves read same 16 rows
    const long xoff = ((long)row * kT + t) * (long)kD;
    const float xv0 = __builtin_nontemporal_load(x + xoff);
    const float xv1 = __builtin_nontemporal_load(x + xoff + 1);
    __builtin_prefetch(x + xoff + (long)16 * kT * kD, 0, 1);

    const _Float16 x0h = (_Float16)xv0;
    const _Float16 x1h = (_Float16)xv1;

    // ---- Layer 1 straight into B-operand regs: H1^T = relu(W1^T x + b1) ----
    v16h bH1[2];
    const v16h zh = {};
    #pragma unroll
    for (int s = 0; s < 2; ++s) {
      v16h h = w1r0[s] * x0h + w1r1[s] * x1h + b1p[s];   // pk_fma chain
      bH1[s] = __builtin_elementwise_max(h, zh);          // pk_max relu
    }

    // ---- Layer 2 + 3: Z^T = W2^T @ H1^T (+b2 via C-init), relu, in-lane W3 dot ----
    v4f a0lo = {}, a0hi = {}, a1lo = {}, a1hi = {};
    #pragma unroll
    for (int j = 0; j < 4; ++j) {
      v8f acc = __builtin_amdgcn_wmma_f32_16x16x32_f16(
          false, aW2[j][0], false, bH1[0], (short)0, b2r[j], false, false);
      acc = __builtin_amdgcn_wmma_f32_16x16x32_f16(
          false, aW2[j][1], false, bH1[1], (short)0, acc, false, false);

      const v8f zf = {};
      const v8f h2 = __builtin_elementwise_max(acc, zf);  // relu
      const v4f h2a = {h2[0], h2[1], h2[2], h2[3]};
      const v4f h2b = {h2[4], h2[5], h2[6], h2[7]};

      const v4f wd0a = *(const v4f*)(w3p0 + 16 * j);       // LDS broadcast loads
      const v4f wd0b = *(const v4f*)(w3p0 + 16 * j + 4);
      const v4f wd1a = *(const v4f*)(w3p1 + 16 * j);
      const v4f wd1b = *(const v4f*)(w3p1 + 16 * j + 4);

      a0lo += h2a * wd0a;  a0hi += h2b * wd0b;             // pk_fma_f32
      a1lo += h2a * wd1a;  a1hi += h2b * wd1b;
    }

    const v4f s0v = a0lo + a0hi;
    const v4f s1v = a1lo + a1hi;
    float s0 = (s0v[0] + s0v[2]) + (s0v[1] + s0v[3]);
    float s1 = (s1v[0] + s1v[2]) + (s1v[1] + s1v[3]);

    // merge partner half (hidden 8..15 of each 16-block) — single exchange
    s0 += __shfl_xor(s0, 16, 32);
    s1 += __shfl_xor(s1, 16, 32);

    const float val = (half ? s1 : s0) + b3v;
    __builtin_nontemporal_store(val, out + ((long)row * kT + t) * (long)kD + half);
  }
}

extern "C" void kernel_launch(void* const* d_in, const int* in_sizes, int n_in,
                              void* d_out, int out_size, void* d_ws, size_t ws_size,
                              hipStream_t stream) {
  (void)n_in; (void)out_size; (void)d_ws; (void)ws_size;
  const float* x  = (const float*)d_in[0];
  const float* W1 = (const float*)d_in[1];
  const float* b1 = (const float*)d_in[2];
  const float* W2 = (const float*)d_in[3];
  const float* b2 = (const float*)d_in[4];
  const float* W3 = (const float*)d_in[5];
  const float* b3 = (const float*)d_in[6];
  float* out = (float*)d_out;

  const int rows = in_sizes[0] / (kT * kD);               // B = 131072
  dim3 grid((unsigned)kT, (unsigned)(rows / ROWS_PER_BLOCK));
  perlead_mlp_wmma_t<<<grid, WAVES * 32, 0, stream>>>(x, W1, b1, W2, b2, W3, b3, out);
}
